// GraphAttentionPooling_16793322128118
// MI455X (gfx1250) — matrix operand order repaired and checked by reading, not compiled, not run on hardware
//
#include <hip/hip_runtime.h>
#include <hip/hip_bf16.h>
#include <math.h>

#define NUM_SEG 1024
#define HID 256
#define HHALF 128          // hidden/2
#define ROWS_PB 128        // rows per block in the score kernel
#define XS_STRIDE 132      // u32 row stride for x tile (16B aligned rows)

typedef __attribute__((ext_vector_type(16))) __bf16 v16bf;
typedef __attribute__((ext_vector_type(8)))  float  v8f;

union FragBF {
  unsigned int u[8];
  uint4 q[2];
  v16bf v;
};

__device__ __forceinline__ unsigned int pack_bf16(float lo, float hi) {
  unsigned int a = __float_as_uint(lo);
  unsigned int b = __float_as_uint(hi);
  a = (a + 0x7FFFu + ((a >> 16) & 1u)) >> 16;   // round-to-nearest-even
  b = (b + 0x7FFFu + ((b >> 16) & 1u)) >> 16;
  return a | (b << 16);
}

// ---------------- K0: pack W1 into bf16 K-pairs, FRAGMENT-MAJOR order --------
// Source: W1 (256x128 f32 row-major).  Logical packed coords: kp=K/2 in [0,128),
// c in [0,128).  B-fragment coords (V_WMMA_F32_16X16X32_BF16, B is 32x16):
//   kt = kp>>4, kpin = kp&15, hi = kpin>>3, v = kpin&7   (VGPR v holds K=2v,2v+1
//   for lanes 0-15; K=16+2v,16+2v+1 for lanes 16-31)
//   nt = c>>4, lane = hi*16 + (c&15)
// dest = ((kt*8+nt)*32 + lane)*8 + v  -> each lane's fragment is 8 contiguous u32.
__global__ void pack_w1_kernel(const float* __restrict__ W1,
                               unsigned int* __restrict__ w1p) {
  for (int i = threadIdx.x; i < HHALF * (HID / 2); i += blockDim.x) {
    int kp = i >> 7, c = i & 127;
    unsigned int p = pack_bf16(W1[(2 * kp) * HHALF + c],
                               W1[(2 * kp + 1) * HHALF + c]);
    int kt = kp >> 4, kpin = kp & 15;
    int hi = kpin >> 3, v = kpin & 7;
    int nt = c >> 4, lane = hi * 16 + (c & 15);
    w1p[(((kt * 8 + nt) * 32 + lane) << 3) + v] = p;
  }
}

// ---------------- K1: scores[r] = tanh(x@W1 + b1) @ W2 + b2 -----------------
__global__ __launch_bounds__(256)
void scores_kernel(const float* __restrict__ x,
                   const unsigned int* __restrict__ w1p,
                   const float* __restrict__ b1,
                   const float* __restrict__ W2,
                   const float* __restrict__ b2,
                   float* __restrict__ scores, int nrows) {
  // x tile: 128 rows x 128 bf16-pairs (u32), 16B-aligned rows
  __shared__ __attribute__((aligned(16))) unsigned int xs[ROWS_PB * XS_STRIDE];
  // W1, fragment-major: 64 fragments x 32 lanes x 8 u32 = 64KB
  __shared__ __attribute__((aligned(16))) unsigned int w1s[64 * 32 * 8];

  const int tid  = threadIdx.x;
  const int lane = tid & 31;
  const int wave = tid >> 5;
  const int l15  = lane & 15;
  const int hi   = lane >> 4;
  const int row0 = blockIdx.x * ROWS_PB;

  // --- Async global->LDS copy of fragment-major W1 (64KB linear copy). ---
  // Each thread issues 16B/lane x 8 iterations: 256 thr * 8 * 16B = 32KB... use 16.
  {
    unsigned lds_base = (unsigned)(unsigned long long)(&w1s[0]);
    #pragma unroll
    for (int it = 0; it < 16; ++it) {
      int idx4 = tid + it * 256;              // uint4 index, 4096 total
      unsigned long long gaddr =
          (unsigned long long)(const void*)(w1p + (idx4 << 2));
      unsigned laddr = lds_base + (unsigned)(idx4 << 4);
      asm volatile("global_load_async_to_lds_b128 %0, %1, off"
                   :: "v"(laddr), "v"(gaddr) : "memory");
    }
  }

  // --- Meanwhile: stage x tile, f32 pairs -> packed bf16 (coalesced). ---
  {
    const float2* x2 = (const float2*)x;
    for (int i = tid; i < ROWS_PB * 128; i += 256) {
      int r = i >> 7, kp = i & 127;
      int gr = row0 + r;
      float2 v = make_float2(0.f, 0.f);
      if (gr < nrows) v = x2[(size_t)gr * 128 + kp];
      xs[r * XS_STRIDE + kp] = pack_bf16(v.x, v.y);
    }
  }

  asm volatile("s_wait_asynccnt 0" ::: "memory");
  __syncthreads();

  // Each wave computes a 16-row strip x all 128 output columns.
  const int mrow = wave * 16;
  const uint4* w1s4 = (const uint4*)w1s;
  const uint4* xrow4 = (const uint4*)&xs[(mrow + l15) * XS_STRIDE];
  v8f acc[8] = {};   // 8 N-tiles of 16x16 f32 accumulators

  #pragma unroll
  for (int kt = 0; kt < 8; ++kt) {           // K = 256 = 8 * 32
    FragBF a;
    // A 16x32 bf16 layout: VGPR 0-3 <- kp = kt*16 + hi*8 + {0..3}
    //                      VGPR 4-7 <- kp = kt*16 + hi*8 + 4 + {0..3}
    a.q[0] = xrow4[kt * 4 + hi * 2];
    a.q[1] = xrow4[kt * 4 + hi * 2 + 1];
    #pragma unroll
    for (int nt = 0; nt < 8; ++nt) {         // N = 128 = 8 * 16
      FragBF bfr;
      int fb = ((kt * 8 + nt) * 32 + lane) * 2;   // uint4 index
      bfr.q[0] = w1s4[fb];
      bfr.q[1] = w1s4[fb + 1];
      acc[nt] = __builtin_amdgcn_wmma_f32_16x16x32_bf16(
          false, a.v, false, bfr.v, (short)0, acc[nt], false, false);
    }
  }

  // Epilogue: per-lane partials over the 8 rows this lane-half owns.
  float part[8] = {0.f, 0.f, 0.f, 0.f, 0.f, 0.f, 0.f, 0.f};
  #pragma unroll
  for (int nt = 0; nt < 8; ++nt) {
    int col = nt * 16 + l15;
    float bb = b1[col];
    float ww = W2[col];
    #pragma unroll
    for (int i = 0; i < 8; ++i)             // C VGPR i -> row i + 8*hi, col
      part[i] += tanhf(acc[nt][i] + bb) * ww;
  }
  // Reduce across the 16 lanes of each half (cols).
  #pragma unroll
  for (int m = 1; m <= 8; m <<= 1) {
    #pragma unroll
    for (int i = 0; i < 8; ++i)
      part[i] += __shfl_xor(part[i], m, 32);
  }
  if (l15 == 0) {
    float bias2 = b2[0];
    #pragma unroll
    for (int i = 0; i < 8; ++i) {
      int r = row0 + mrow + hi * 8 + i;
      if (r < nrows) scores[r] = part[i] + bias2;
    }
  }
}

// ---------------- K2: segment boundaries via binary search (batch sorted) ----
__global__ void segstart_kernel(const int* __restrict__ batch, int n,
                                int* __restrict__ segs) {
  int sid = blockIdx.x * blockDim.x + threadIdx.x;
  if (sid > NUM_SEG) return;
  int lo = 0, hicnt = n;
  while (lo < hicnt) {
    int mid = (lo + hicnt) >> 1;
    if (batch[mid] < sid) lo = mid + 1; else hicnt = mid;
  }
  segs[sid] = lo;   // segs[1024] == n
}

// ---------------- K3: per-segment softmax + weighted pooling -----------------
__global__ __launch_bounds__(256)
void pool_kernel(const float* __restrict__ x,
                 const float* __restrict__ scores,
                 const int* __restrict__ segs,
                 float* __restrict__ out) {
  __shared__ float red[256];
  __shared__ float wl[256];
  const int tid = threadIdx.x;
  const int seg = blockIdx.x;
  const int s0 = segs[seg], s1 = segs[seg + 1];

  // 1) segment max
  float m = -INFINITY;
  for (int r = s0 + tid; r < s1; r += 256) m = fmaxf(m, scores[r]);
  red[tid] = m; __syncthreads();
  for (int off = 128; off > 0; off >>= 1) {
    if (tid < off) red[tid] = fmaxf(red[tid], red[tid + off]);
    __syncthreads();
  }
  m = red[0]; __syncthreads();

  // 2) segment sum of exp
  float sm = 0.f;
  for (int r = s0 + tid; r < s1; r += 256) sm += __expf(scores[r] - m);
  red[tid] = sm; __syncthreads();
  for (int off = 128; off > 0; off >>= 1) {
    if (tid < off) red[tid] += red[tid + off];
    __syncthreads();
  }
  const float denom = red[0] + 1e-16f;

  // 3) weighted column accumulation; thread == column
  float acc = 0.f;
  for (int base = s0; base < s1; base += 256) {
    __syncthreads();
    int r = base + tid;
    wl[tid] = (r < s1) ? __expf(scores[r] - m) / denom : 0.f;
    __syncthreads();
    int cnt = min(256, s1 - base);
    #pragma unroll 8
    for (int j = 0; j < cnt; ++j)
      acc += x[(size_t)(base + j) * HID + tid] * wl[j];
  }
  out[seg * HID + tid] = acc;   // empty segments write 0 (matches segment_sum)
}

// -----------------------------------------------------------------------------
extern "C" void kernel_launch(void* const* d_in, const int* in_sizes, int n_in,
                              void* d_out, int out_size, void* d_ws, size_t ws_size,
                              hipStream_t stream) {
  const float* x    = (const float*)d_in[0];
  const int*   batch= (const int*)d_in[1];
  const float* W1   = (const float*)d_in[2];
  const float* b1   = (const float*)d_in[3];
  const float* W2   = (const float*)d_in[4];
  const float* b2   = (const float*)d_in[5];
  const int nrows   = in_sizes[1];          // 1,000,000

  // workspace layout: scores (nrows f32) | segs (1025 i32, padded) | w1p (16384 u32)
  float* scores = (float*)d_ws;
  size_t off = (size_t)nrows * sizeof(float);
  int* segs = (int*)((char*)d_ws + off);
  off += ((NUM_SEG + 1) * sizeof(int) + 15) & ~(size_t)15;
  unsigned int* w1p = (unsigned int*)((char*)d_ws + ((off + 15) & ~(size_t)15));

  pack_w1_kernel<<<1, 256, 0, stream>>>(W1, w1p);
  scores_kernel<<<(nrows + ROWS_PB - 1) / ROWS_PB, 256, 0, stream>>>(
      x, w1p, b1, W2, b2, scores, nrows);
  segstart_kernel<<<(NUM_SEG + 256) / 256, 256, 0, stream>>>(batch, nrows, segs);
  pool_kernel<<<NUM_SEG, 256, 0, stream>>>(x, scores, segs, (float*)d_out);
}
